// DilatedAttention_88862873354265
// MI455X (gfx1250) — matrix operand order repaired and checked by reading, not compiled, not run on hardware
//
#include <hip/hip_runtime.h>
#include <math.h>

// ---------------------------------------------------------------- constants
#define DIMX      512
#define NHEADS    8
#define HDIM      64
#define BATCH     2
#define SEQ       8192
#define MTOT      (BATCH * SEQ)       // 16384 rows for the big GEMMs
#define LSUB      1024                // gathered sub-sequence length (all branches)
#define SCALEF    0.125f              // 64^-0.5
#define NEG_FILL  (-100000000.0f)

typedef __bf16 bf16;
typedef __attribute__((ext_vector_type(16))) __bf16 v16bf;
typedef __attribute__((ext_vector_type(8)))  float  v8f;

__device__ __forceinline__ float bf2f(bf16 x) { return (float)x; }
__device__ __forceinline__ bf16  f2bf(float x) { return (bf16)x; }

union Pack8 { uint4 u; bf16 h[8]; };           // 8 bf16 <-> 16 bytes
union Pack4 { uint2 u; bf16 h[4]; };           // 4 bf16 <-> 8 bytes

// -------------------------------------------------- WMMA fragment helpers
// A fragment: 16x32 bf16, row-major source with row stride `ld` (elements).
// lanes 0-15 -> M=lane, K={0..7,16..23}; lanes 16-31 -> M=lane-16, K={8..15,24..31}.
__device__ __forceinline__ v16bf load_frag_a(const bf16* p, int ld) {
  const int lane = threadIdx.x & 31;
  const int m    = lane & 15;
  const int kg   = (lane >> 4) * 8;
  const bf16* row = p + m * ld;
  v16bf a;
#pragma unroll
  for (int j = 0; j < 4; ++j) {
    a[2 * j]     = row[kg + 2 * j];
    a[2 * j + 1] = row[kg + 2 * j + 1];
    a[8 + 2 * j]     = row[16 + kg + 2 * j];
    a[8 + 2 * j + 1] = row[16 + kg + 2 * j + 1];
  }
  return a;
}

// B fragment: 32x16 (KxN) from an N-major (N rows x 32 K) tile, stride `ld`.
// lanes 0-15 -> N=lane, K={0..7,16..23}; lanes 16-31 -> N=lane-16, K={8..15,24..31}.
__device__ __forceinline__ v16bf load_frag_b(const bf16* p, int ld) {
  const int lane = threadIdx.x & 31;
  const int n    = lane & 15;
  const int kg   = (lane >> 4) * 8;
  const bf16* row = p + n * ld;
  v16bf b;
#pragma unroll
  for (int j = 0; j < 8; ++j) {
    b[j]     = row[kg + j];
    b[8 + j] = row[16 + kg + j];
  }
  return b;
}

// ---------------------------------------------------------- weight convert
__global__ __launch_bounds__(256) void cvt_w(const float* a, const float* b,
                                             const float* c, const float* d,
                                             bf16* oa, bf16* ob, bf16* oc,
                                             bf16* od) {
  int t = blockIdx.x * 256 + threadIdx.x;  // over DIMX*DIMX/4
  if (t < DIMX * DIMX / 4) {
    const float* srcs[4] = {a, b, c, d};
    bf16* dsts[4] = {oa, ob, oc, od};
#pragma unroll
    for (int i = 0; i < 4; ++i) {
      float4 v = *(const float4*)(srcs[i] + 4 * (size_t)t);
      Pack4 p;
      p.h[0] = f2bf(v.x); p.h[1] = f2bf(v.y);
      p.h[2] = f2bf(v.z); p.h[3] = f2bf(v.w);
      *(uint2*)(dsts[i] + 4 * (size_t)t) = p.u;
    }
  }
}

// -------------------------------------------------------------- GEMM: C = A @ W^T
// A: (M,K) f32 or bf16, W: (N,K) bf16 row-major, C: (M,N) bf16 or f32(+bias).
// Block tile 64(M) x 128(N); 8 waves, each wave a 32x32 sub-tile (2x2 WMMA).
template <bool A_F32, bool OUT_F32>
__global__ __launch_bounds__(256) void gemm_wmma(const void* __restrict__ Ap,
                                                 const bf16* __restrict__ W,
                                                 const float* __restrict__ bias,
                                                 void* __restrict__ Cp,
                                                 int M, int N, int K) {
  __shared__ bf16 As[64 * 40];    // 64x32, stride 40 (80 B rows, 16 B-aligned)
  __shared__ bf16 Bs[128 * 40];   // 128x32, stride 40
  const int tid  = threadIdx.x;
  const int m0   = blockIdx.x * 64;
  const int n0   = blockIdx.y * 128;
  const int wave = tid >> 5;
  const int wm   = (wave & 1) * 32;
  const int wn   = (wave >> 1) * 32;
  const float* Af = (const float*)Ap;
  const bf16*  Ab = (const bf16*)Ap;

  v8f acc[2][2];
#pragma unroll
  for (int i = 0; i < 2; ++i)
#pragma unroll
    for (int j = 0; j < 2; ++j)
#pragma unroll
      for (int r = 0; r < 8; ++r) acc[i][j][r] = 0.0f;

  for (int k0 = 0; k0 < K; k0 += 32) {
    if (A_F32) {
      // A tile 64x32 f32 -> bf16: 512 float4 loads / block
#pragma unroll
      for (int i = 0; i < 2; ++i) {
        int idx = tid + i * 256;
        int r = idx >> 3, c4 = (idx & 7) * 4;
        float4 v = *(const float4*)(Af + (size_t)(m0 + r) * K + k0 + c4);
        Pack4 p;
        p.h[0] = f2bf(v.x); p.h[1] = f2bf(v.y);
        p.h[2] = f2bf(v.z); p.h[3] = f2bf(v.w);
        *(uint2*)(As + r * 40 + c4) = p.u;
      }
    } else {
      // A tile 64x32 bf16: 256 uint4 loads / block
      int r = tid >> 2, c8 = (tid & 3) * 8;
      *(uint4*)(As + r * 40 + c8) =
          *(const uint4*)(Ab + (size_t)(m0 + r) * K + k0 + c8);
    }
    // B tile 128x32 bf16: 512 uint4 loads / block
#pragma unroll
    for (int i = 0; i < 2; ++i) {
      int idx = tid + i * 256;
      int r = idx >> 2, c8 = (idx & 3) * 8;
      *(uint4*)(Bs + r * 40 + c8) =
          *(const uint4*)(W + (size_t)(n0 + r) * K + k0 + c8);
    }
    __syncthreads();

    v16bf a0 = load_frag_a(As + wm * 40, 40);
    v16bf a1 = load_frag_a(As + (wm + 16) * 40, 40);
    v16bf b0 = load_frag_b(Bs + wn * 40, 40);
    v16bf b1 = load_frag_b(Bs + (wn + 16) * 40, 40);
    acc[0][0] = __builtin_amdgcn_wmma_f32_16x16x32_bf16(false, a0, false, b0,
                                                        (short)0, acc[0][0],
                                                        false, false);
    acc[0][1] = __builtin_amdgcn_wmma_f32_16x16x32_bf16(false, a0, false, b1,
                                                        (short)0, acc[0][1],
                                                        false, false);
    acc[1][0] = __builtin_amdgcn_wmma_f32_16x16x32_bf16(false, a1, false, b0,
                                                        (short)0, acc[1][0],
                                                        false, false);
    acc[1][1] = __builtin_amdgcn_wmma_f32_16x16x32_bf16(false, a1, false, b1,
                                                        (short)0, acc[1][1],
                                                        false, false);
    __syncthreads();
  }

  // C layout: VGPR r -> (M = mb + r, N = lane&15), mb = 8*(lane>>4)
  const int lane = tid & 31;
  const int cn   = lane & 15;
  const int mb   = (lane >> 4) * 8;
#pragma unroll
  for (int i = 0; i < 2; ++i)
#pragma unroll
    for (int j = 0; j < 2; ++j) {
      int row0 = m0 + wm + 16 * i + mb;
      int col  = n0 + wn + 16 * j + cn;
#pragma unroll
      for (int r = 0; r < 8; ++r) {
        float v = acc[i][j][r];
        if (OUT_F32) {
          ((float*)Cp)[(size_t)(row0 + r) * N + col] =
              v + (bias ? bias[col] : 0.0f);
        } else {
          ((bf16*)Cp)[(size_t)(row0 + r) * N + col] = f2bf(v);
        }
      }
    }
}

// ------------------------------------------------------- dilated attention
// One block = (batch, segment, head, 128-row Q chunk) of a gathered
// 1024x1024x64 attention problem. Flash attention with online softmax.
__global__ __launch_bounds__(256) void attn_wmma(const bf16* __restrict__ Q,
                                                 const bf16* __restrict__ K,
                                                 const bf16* __restrict__ V,
                                                 bf16* __restrict__ O,
                                                 float* __restrict__ LSE,
                                                 int sl, int dr, int nseg) {
  __shared__ bf16 Qs[128 * 72];  // 128 q-rows x 64 d (144 B rows)
  __shared__ bf16 Ks[64 * 72];   // 64 k-rows x 64 d
  __shared__ bf16 Vt[64 * 72];   // transposed: 64 d-rows x 64 keys
  __shared__ bf16 Ps[128 * 72];  // P staging + output staging

  const int tid  = threadIdx.x;
  const int qb   = blockIdx.x;             // 0..7
  const int seg  = blockIdx.y / NHEADS;
  const int h    = blockIdx.y % NHEADS;
  const int bidx = blockIdx.z;
  const int hsub = NHEADS / dr;
  const int offp = h / hsub;               // per-head dilation offset
  const int qrow0 = qb * 128;

  // gathered element l' lives at dense position seg*sl + l'*dr + offp
  const size_t densebase =
      ((size_t)bidx * SEQ + (size_t)seg * sl + offp) * DIMX + (size_t)h * HDIM;

  // Q tile: 1024 x uint4 (8 bf16) loads
#pragma unroll
  for (int i = 0; i < 4; ++i) {
    int idx = tid + i * 256;
    int r = idx >> 3, dv = (idx & 7) * 8;
    *(uint4*)(Qs + r * 72 + dv) =
        *(const uint4*)(Q + densebase + (size_t)(qrow0 + r) * dr * DIMX + dv);
  }
  __syncthreads();

  const int wave = tid >> 5;
  const int lane = tid & 31;
  const int mb   = (lane >> 4) * 8;
  const int cn   = lane & 15;

  float mrow[8], lrow[8];
#pragma unroll
  for (int r = 0; r < 8; ++r) { mrow[r] = -1e30f; lrow[r] = 0.0f; }
  v8f oacc[4];
#pragma unroll
  for (int dt = 0; dt < 4; ++dt)
#pragma unroll
    for (int r = 0; r < 8; ++r) oacc[dt][r] = 0.0f;

  for (int kc = 0; kc < LSUB; kc += 64) {
    // K + V chunk: 64x64 each, vectorized 16 B loads
#pragma unroll
    for (int i = 0; i < 2; ++i) {
      int idx = tid + i * 256;
      int r = idx >> 3, dv = (idx & 7) * 8;
      size_t g = densebase + (size_t)(kc + r) * dr * DIMX + dv;
      *(uint4*)(Ks + r * 72 + dv) = *(const uint4*)(K + g);
      Pack8 pv;
      pv.u = *(const uint4*)(V + g);
#pragma unroll
      for (int j = 0; j < 8; ++j) Vt[(dv + j) * 72 + r] = pv.h[j];  // transpose
      // prefetch next chunk into cache (global_prefetch_b8)
      if (kc + 64 < LSUB) {
        size_t gn = densebase + (size_t)(kc + 64 + r) * dr * DIMX + dv;
        __builtin_prefetch((const void*)(K + gn), 0, 0);
        __builtin_prefetch((const void*)(V + gn), 0, 0);
      }
    }
    __syncthreads();

    // S = Q @ K^T  (16 q-rows per wave x 64 keys)
    v8f s[4];
#pragma unroll
    for (int nt = 0; nt < 4; ++nt)
#pragma unroll
      for (int r = 0; r < 8; ++r) s[nt][r] = 0.0f;
    const bf16* qrow = Qs + wave * 16 * 72;
#pragma unroll
    for (int kk = 0; kk < 2; ++kk) {
      v16bf a = load_frag_a(qrow + kk * 32, 72);
#pragma unroll
      for (int nt = 0; nt < 4; ++nt) {
        v16bf b = load_frag_b(Ks + nt * 16 * 72 + kk * 32, 72);
        s[nt] = __builtin_amdgcn_wmma_f32_16x16x32_bf16(false, a, false, b,
                                                        (short)0, s[nt],
                                                        false, false);
      }
    }

    // online softmax over this 64-key chunk
    float mnew[8];
#pragma unroll
    for (int r = 0; r < 8; ++r) {
      float mx = -1e30f;
#pragma unroll
      for (int nt = 0; nt < 4; ++nt) mx = fmaxf(mx, s[nt][r]);
#pragma unroll
      for (int m = 1; m < 16; m <<= 1) mx = fmaxf(mx, __shfl_xor(mx, m, 32));
      mnew[r] = fmaxf(mrow[r], mx * SCALEF);
    }
#pragma unroll
    for (int r = 0; r < 8; ++r) {
      float rs = 0.0f;
#pragma unroll
      for (int nt = 0; nt < 4; ++nt) {
        float p = __expf(s[nt][r] * SCALEF - mnew[r]);
        rs += p;
        Ps[(wave * 16 + mb + r) * 72 + nt * 16 + cn] = f2bf(p);
      }
#pragma unroll
      for (int m = 1; m < 16; m <<= 1) rs += __shfl_xor(rs, m, 32);
      float alpha = __expf(mrow[r] - mnew[r]);
      lrow[r] = lrow[r] * alpha + rs;
      mrow[r] = mnew[r];
#pragma unroll
      for (int dt = 0; dt < 4; ++dt) oacc[dt][r] *= alpha;
    }

    // O += P @ V
    const bf16* prow = Ps + wave * 16 * 72;
#pragma unroll
    for (int kk = 0; kk < 2; ++kk) {
      v16bf a = load_frag_a(prow + kk * 32, 72);
#pragma unroll
      for (int dt = 0; dt < 4; ++dt) {
        v16bf b = load_frag_b(Vt + dt * 16 * 72 + kk * 32, 72);
        oacc[dt] = __builtin_amdgcn_wmma_f32_16x16x32_bf16(false, a, false, b,
                                                           (short)0, oacc[dt],
                                                           false, false);
      }
    }
    __syncthreads();
  }

  // epilogue: normalize, stage through LDS, vectorized global stores
  const size_t rowbase =
      (((size_t)bidx * nseg + seg) * NHEADS + h) * LSUB + qrow0 + wave * 16;
  const size_t obase = rowbase * HDIM;
#pragma unroll
  for (int r = 0; r < 8; ++r) {
    float inv = 1.0f / lrow[r];
#pragma unroll
    for (int dt = 0; dt < 4; ++dt)
      Ps[(wave * 16 + mb + r) * 72 + dt * 16 + cn] = f2bf(oacc[dt][r] * inv);
  }
#pragma unroll
  for (int i = 0; i < 4; ++i) {
    int vi = lane + 32 * i;            // 0..127 -> 16 rows x 8 d-groups
    int row = vi >> 3, dg = (vi & 7) * 8;
    *(uint4*)(O + obase + (size_t)row * HDIM + dg) =
        *(const uint4*)(Ps + (wave * 16 + row) * 72 + dg);
  }
  if (cn == 0) {
#pragma unroll
    for (int r = 0; r < 8; ++r)
      LSE[rowbase + mb + r] = mrow[r] + __logf(lrow[r]);
  }
}

// -------------------------------------------------- branch combination
__global__ __launch_bounds__(256) void combine_kernel(
    const bf16* __restrict__ O1, const float* __restrict__ L1,
    const bf16* __restrict__ O2, const float* __restrict__ L2,
    const bf16* __restrict__ O3, const float* __restrict__ L3,
    bf16* __restrict__ X) {
  int t = blockIdx.x * 256 + threadIdx.x;  // over b*L*H
  if (t >= BATCH * SEQ * NHEADS) return;
  const int h = t % NHEADS;
  const int l = (t / NHEADS) % SEQ;
  const int b = t / (NHEADS * SEQ);

  const int sls[3] = {1024, 2048, 4096};
  const int drs[3] = {1, 2, 4};
  const int nss[3] = {8, 4, 2};
  const bf16* Os[3] = {O1, O2, O3};
  const float* Ls[3] = {L1, L2, L3};

  float lse[3];
  size_t ob[3];
  bool cov[3];
#pragma unroll
  for (int i = 0; i < 3; ++i) {
    int sl = sls[i], dr = drs[i];
    int seg = l / sl, s = l % sl;
    int offp = h / (NHEADS / dr);
    bool c = (s % dr) == offp;
    int lp = (s - offp) / dr;
    size_t base = (((size_t)b * nss[i] + seg) * NHEADS + h) * LSUB + lp;
    cov[i] = c;
    lse[i] = c ? Ls[i][base] : NEG_FILL;
    ob[i] = base * HDIM;
  }
  float m = fmaxf(lse[0], fmaxf(lse[1], lse[2]));
  float w[3], ws = 0.0f;
#pragma unroll
  for (int i = 0; i < 3; ++i) { w[i] = __expf(lse[i] - m); ws += w[i]; }
  float inv = 1.0f / ws;
#pragma unroll
  for (int i = 0; i < 3; ++i) w[i] *= inv;

  bf16* xp = X + ((size_t)b * SEQ + l) * DIMX + (size_t)h * HDIM;
#pragma unroll
  for (int dg = 0; dg < HDIM; dg += 8) {
    float acc[8];
#pragma unroll
    for (int j = 0; j < 8; ++j) acc[j] = 0.0f;
#pragma unroll
    for (int i = 0; i < 3; ++i)
      if (cov[i]) {
        Pack8 p;
        p.u = *(const uint4*)(Os[i] + ob[i] + dg);
#pragma unroll
        for (int j = 0; j < 8; ++j) acc[j] += w[i] * bf2f(p.h[j]);
      }
    Pack8 o;
#pragma unroll
    for (int j = 0; j < 8; ++j) o.h[j] = f2bf(acc[j]);
    *(uint4*)(xp + dg) = o.u;
  }
}

// ------------------------------------------------------------ host launch
extern "C" void kernel_launch(void* const* d_in, const int* in_sizes, int n_in,
                              void* d_out, int out_size, void* d_ws,
                              size_t ws_size, hipStream_t stream) {
  (void)in_sizes; (void)n_in; (void)out_size; (void)ws_size;
  const float* query = (const float*)d_in[0];
  const float* key_  = (const float*)d_in[1];
  const float* value = (const float*)d_in[2];
  const float* Wq    = (const float*)d_in[3];
  const float* Wk    = (const float*)d_in[4];
  const float* Wv    = (const float*)d_in[5];
  const float* Wo    = (const float*)d_in[6];
  const float* bo    = (const float*)d_in[7];
  float* out = (float*)d_out;

  char* ws = (char*)d_ws;
  size_t off = 0;
  auto alloc = [&](size_t bytes) -> void* {
    void* p = (void*)(ws + off);
    off += (bytes + 255) & ~(size_t)255;
    return p;
  };
  bf16* Wqb = (bf16*)alloc((size_t)DIMX * DIMX * 2);
  bf16* Wkb = (bf16*)alloc((size_t)DIMX * DIMX * 2);
  bf16* Wvb = (bf16*)alloc((size_t)DIMX * DIMX * 2);
  bf16* Wob = (bf16*)alloc((size_t)DIMX * DIMX * 2);
  bf16* Qb  = (bf16*)alloc((size_t)MTOT * DIMX * 2);
  bf16* Kb  = (bf16*)alloc((size_t)MTOT * DIMX * 2);
  bf16* Vb  = (bf16*)alloc((size_t)MTOT * DIMX * 2);
  bf16* Xb  = (bf16*)alloc((size_t)MTOT * DIMX * 2);
  bf16* O1  = (bf16*)alloc((size_t)BATCH * 8 * NHEADS * LSUB * HDIM * 2);
  bf16* O2  = (bf16*)alloc((size_t)BATCH * 4 * NHEADS * LSUB * HDIM * 2);
  bf16* O3  = (bf16*)alloc((size_t)BATCH * 2 * NHEADS * LSUB * HDIM * 2);
  float* L1 = (float*)alloc((size_t)BATCH * 8 * NHEADS * LSUB * 4);
  float* L2 = (float*)alloc((size_t)BATCH * 4 * NHEADS * LSUB * 4);
  float* L3 = (float*)alloc((size_t)BATCH * 2 * NHEADS * LSUB * 4);

  // 1) weights -> bf16
  cvt_w<<<dim3((DIMX * DIMX / 4 + 255) / 256), dim3(256), 0, stream>>>(
      Wq, Wk, Wv, Wo, Wqb, Wkb, Wvb, Wob);

  // 2) projections (f32 A, bf16 out)
  dim3 ggrid(MTOT / 64, DIMX / 128);
  gemm_wmma<true, false><<<ggrid, 256, 0, stream>>>(query, Wqb, nullptr, Qb,
                                                    MTOT, DIMX, DIMX);
  gemm_wmma<true, false><<<ggrid, 256, 0, stream>>>(key_, Wkb, nullptr, Kb,
                                                    MTOT, DIMX, DIMX);
  gemm_wmma<true, false><<<ggrid, 256, 0, stream>>>(value, Wvb, nullptr, Vb,
                                                    MTOT, DIMX, DIMX);

  // 3) attention branches: (sl, dr, nseg) = (1024,1,8) (2048,2,4) (4096,4,2)
  attn_wmma<<<dim3(8, 8 * NHEADS, BATCH), 256, 0, stream>>>(Qb, Kb, Vb, O1, L1,
                                                            1024, 1, 8);
  attn_wmma<<<dim3(8, 4 * NHEADS, BATCH), 256, 0, stream>>>(Qb, Kb, Vb, O2, L2,
                                                            2048, 2, 4);
  attn_wmma<<<dim3(8, 2 * NHEADS, BATCH), 256, 0, stream>>>(Qb, Kb, Vb, O3, L3,
                                                            4096, 4, 2);

  // 4) LSE-weighted branch combination -> dense bf16 (b, L, 512)
  combine_kernel<<<dim3((BATCH * SEQ * NHEADS + 255) / 256), dim3(256), 0,
                   stream>>>(O1, L1, O2, L2, O3, L3, Xb);

  // 5) output projection (bf16 A, f32 out + bias)
  gemm_wmma<false, true><<<ggrid, 256, 0, stream>>>(Xb, Wob, bo, out, MTOT,
                                                    DIMX, DIMX);
}
